// ConnectomeHybridActor_5085241279254
// MI455X (gfx1250) — compile-verified, matrix operand order checked by reading, not run on hardware
//
#include <hip/hip_runtime.h>

typedef __attribute__((ext_vector_type(16))) __bf16 v16bf;
typedef __attribute__((ext_vector_type(8)))  __bf16 v8bf;
typedef __attribute__((ext_vector_type(8)))  float  v8f;

// ---- dims ----
#define BSZ   1024
#define TT    32
#define OBSD  256
#define NORN  341
#define NPN   115
#define NKC   454
#define NLN   114
// padded dims (multiple of 32 for K, 16 for N)
#define PORN  352
#define PPN   128
#define PKC   480
#define PLN   128
// n-tiles (padded/16)
#define TORN  22
#define TPN   8
#define TKC   30
#define TLN   8
// k-tiles (padded/32)
#define KT_ORN 11
#define KT_PN  4
#define KT_KC  15
#define KT_LN  4
#define KT_OBS 8

#define MROWS 32          // batch rows per workgroup (2 WMMA M-tiles)
#define TILE_E 512        // bf16 elements per packed 32x16 B tile

// packed-weight offsets in workspace (bf16 elements)
constexpr size_t O_INW = 0;
constexpr size_t O_PTO = O_INW + (size_t)8  * 22 * TILE_E;
constexpr size_t O_OTO = O_PTO + (size_t)4  * 22 * TILE_E;
constexpr size_t O_LTO = O_OTO + (size_t)11 * 22 * TILE_E;
constexpr size_t O_OTP = O_LTO + (size_t)4  * 22 * TILE_E;
constexpr size_t O_KTP = O_OTP + (size_t)11 * 8  * TILE_E;
constexpr size_t O_LTP = O_KTP + (size_t)15 * 8  * TILE_E;
constexpr size_t O_PTP = O_LTP + (size_t)4  * 8  * TILE_E;
constexpr size_t O_PTL = O_PTP + (size_t)4  * 8  * TILE_E;
constexpr size_t O_OTL = O_PTL + (size_t)4  * 8  * TILE_E;
constexpr size_t O_LTL = O_OTL + (size_t)11 * 8  * TILE_E;
constexpr size_t O_PTK = O_LTL + (size_t)4  * 8  * TILE_E;
constexpr size_t O_KTK = O_PTK + (size_t)4  * 30 * TILE_E;
constexpr size_t O_END = O_KTK + (size_t)15 * 30 * TILE_E;

// output offsets (f32 elements)
constexpr size_t OUT_MU = 0;
constexpr size_t OUT_LS = (size_t)BSZ * TT * 2;          // 65536
constexpr size_t OUT_C  = OUT_LS + (size_t)BSZ * TT * 2; // 131072
constexpr size_t OUT_H2 = OUT_C + (size_t)BSZ * TT;      // 163840

// ---------------------------------------------------------------------------
// Pack one (K,N) f32 weight matrix into WMMA-B 32x16 bf16 tile fragments.
// ---------------------------------------------------------------------------
__global__ void pack_w_kernel(const float* __restrict__ src, __bf16* __restrict__ dst,
                              int kact, int nact, int ntiles) {
  int tile = blockIdx.x;
  int kt = tile / ntiles, nt = tile % ntiles;
  for (int e = threadIdx.x; e < TILE_E; e += blockDim.x) {
    int l = e >> 4, i = e & 15;
    int kb = (l >> 4) * 8;
    int k  = kt * 32 + ((i < 8) ? (kb + i) : (16 + kb + (i - 8)));
    int n  = nt * 16 + (l & 15);
    float v = (k < kact && n < nact) ? src[(size_t)k * nact + n] : 0.0f;
    dst[(size_t)tile * TILE_E + e] = (__bf16)v;
  }
}

// ---------------------------------------------------------------------------
// Fold readout + heads:  headW[k][oc] = sum_h R_W[k][h] * headcol(h,oc)
// ---------------------------------------------------------------------------
__global__ void fold_heads_kernel(const float* __restrict__ RW, const float* __restrict__ Rb,
                                  const float* __restrict__ muW, const float* __restrict__ mub,
                                  const float* __restrict__ lsW, const float* __restrict__ lsb,
                                  const float* __restrict__ cW,  const float* __restrict__ cb,
                                  float* __restrict__ headW, float* __restrict__ headB) {
  int idx = blockIdx.x * blockDim.x + threadIdx.x;
  if (idx < NKC * 5) {
    int k = idx / 5, oc = idx % 5;
    float s = 0.f;
    for (int h = 0; h < 1024; ++h) {
      float w = (oc < 2) ? muW[h * 2 + oc] : (oc < 4) ? lsW[h * 2 + (oc - 2)] : cW[h];
      s += RW[(size_t)k * 1024 + h] * w;
    }
    headW[idx] = s;
  } else if (idx < NKC * 5 + 5) {
    int oc = idx - NKC * 5;
    float s = (oc < 2) ? mub[oc] : (oc < 4) ? lsb[oc - 2] : cb[0];
    for (int h = 0; h < 1024; ++h) {
      float w = (oc < 2) ? muW[h * 2 + oc] : (oc < 4) ? lsW[h * 2 + (oc - 2)] : cW[h];
      s += Rb[h] * w;
    }
    headB[oc] = s;
  }
}

// ---------------------------------------------------------------------------
// Device helpers
// ---------------------------------------------------------------------------
__device__ __forceinline__ v8f wmma_bf16(v16bf a, v16bf b, v8f c) {
  return __builtin_amdgcn_wmma_f32_16x16x32_bf16(false, a, false, b, (short)0, c,
                                                 false, false);
}

// A fragment (16x32 bf16 activations) for M-tile m from LDS row-major state.
__device__ __forceinline__ v16bf afrag_lds(const __bf16* s, int stride, int kt, int lane,
                                           int m) {
  int row = m * 16 + (lane & 15);
  int kb  = (lane >> 4) * 8;
  const __bf16* p = s + row * stride + kt * 32 + kb;
  v8bf lo = *(const v8bf*)p;
  v8bf hi = *(const v8bf*)(p + 16);
  v16bf a;
#pragma unroll
  for (int i = 0; i < 8; ++i) { a[i] = lo[i]; a[8 + i] = hi[i]; }
  return a;
}

// B fragment (32x16 bf16 weights) from packed global tiles: one 32B load/lane.
__device__ __forceinline__ v16bf bfrag_gl(const __bf16* w, int ntiles, int kt, int nt,
                                          int lane) {
  return *(const v16bf*)(w + ((size_t)(kt * ntiles + nt)) * TILE_E + lane * 16);
}

// Dual-M GEMM: one B fragment feeds two independent WMMAs (M-tiles 0 and 1).
__device__ __forceinline__ void gemm_acc2(v8f& a0, v8f& a1, const __bf16* s, int stride,
                                          int ktiles, const __bf16* w, int ntiles, int nt,
                                          int lane) {
  __builtin_prefetch(w + ((size_t)nt) * TILE_E, 0, 1);
#pragma unroll
  for (int kt = 0; kt < 16; ++kt) {
    if (kt >= ktiles) break;
    v16bf b  = bfrag_gl(w, ntiles, kt, nt, lane);
    v16bf f0 = afrag_lds(s, stride, kt, lane, 0);
    v16bf f1 = afrag_lds(s, stride, kt, lane, 1);
    a0 = wmma_bf16(f0, b, a0);
    a1 = wmma_bf16(f1, b, a1);
  }
}

__device__ __forceinline__ void store_tile(__bf16* s, int stride, int nt, int nact, v8f v,
                                           int lane, int m) {
  int col   = nt * 16 + (lane & 15);
  int rbase = m * 16 + (lane >> 4) * 8;
#pragma unroll
  for (int j = 0; j < 8; ++j) {
    float x = (col < nact) ? v[j] : 0.0f;
    s[(rbase + j) * stride + col] = (__bf16)x;
  }
}

__device__ __forceinline__ float fast_tanh(float x) {
  float cl = x > 10.f ? 10.f : (x < -10.f ? -10.f : x);
  float e  = __builtin_amdgcn_exp2f(cl * 2.885390082f);   // exp(2x) = exp2(2x*log2 e)
  return 1.f - 2.f * __builtin_amdgcn_rcpf(e + 1.f);
}

// ---------------------------------------------------------------------------
// Persistent recurrence: 1 block = 32 batch rows (2 M-tiles), 8 waves.
// State lives in LDS (bf16); weights stream from L2 as packed WMMA-B frags,
// each fragment reused by two WMMAs.
// ---------------------------------------------------------------------------
__global__ void __launch_bounds__(256)
connectome_recurrence(const __bf16* __restrict__ wp,
                      const float* __restrict__ headW, const float* __restrict__ headB,
                      const float* __restrict__ obs,   const float* __restrict__ in_b,
                      const float* __restrict__ b_orn, const float* __restrict__ b_pn,
                      const float* __restrict__ b_kc,  const float* __restrict__ b_ln,
                      float* __restrict__ out) {
  __shared__ __bf16 sORN[MROWS * PORN];
  __shared__ __bf16 sPN [MROWS * PPN];
  __shared__ __bf16 sKC [MROWS * PKC];
  __shared__ __bf16 sLN [MROWS * PLN];
  __shared__ float  sX  [MROWS * PORN];

  const int tid   = threadIdx.x;
  const int lane  = tid & 31;
  const int wave  = tid >> 5;
  const int nlane = lane & 15;
  const int half  = lane >> 4;
  const int rbase = half * 8;
  const int b0    = blockIdx.x * MROWS;

  const __bf16* Winw = wp + O_INW;
  const __bf16* Wpto = wp + O_PTO;
  const __bf16* Woto = wp + O_OTO;
  const __bf16* Wlto = wp + O_LTO;
  const __bf16* Wotp = wp + O_OTP;
  const __bf16* Wktp = wp + O_KTP;
  const __bf16* Wltp = wp + O_LTP;
  const __bf16* Wptp = wp + O_PTP;
  const __bf16* Wptl = wp + O_PTL;
  const __bf16* Wotl = wp + O_OTL;
  const __bf16* Wltl = wp + O_LTL;
  const __bf16* Wptk = wp + O_PTK;
  const __bf16* Wktk = wp + O_KTK;

  // zero state
  for (int i = tid; i < MROWS * PORN; i += 256) sORN[i] = (__bf16)0.f;
  for (int i = tid; i < MROWS * PPN;  i += 256) sPN[i]  = (__bf16)0.f;
  for (int i = tid; i < MROWS * PKC;  i += 256) sKC[i]  = (__bf16)0.f;
  for (int i = tid; i < MROWS * PLN;  i += 256) sLN[i]  = (__bf16)0.f;
  __syncthreads();

#pragma unroll 1
  for (int t = 0; t < TT; ++t) {
    // ---- x_t = obs[:, t, :] @ in_W + in_b   (WMMA, B frag shared by both M) ----
#pragma unroll
    for (int ti = 0; ti < 3; ++ti) {
      int nt = wave + ti * 8;
      if (nt < TORN) {
        v8f acc0 = {}, acc1 = {};
        int kb = half * 8;
#pragma unroll
        for (int kt = 0; kt < KT_OBS; ++kt) {
          v16bf b = bfrag_gl(Winw, TORN, kt, nt, lane);
#pragma unroll
          for (int m = 0; m < 2; ++m) {
            int row = m * 16 + nlane;
            const float* p = obs + ((size_t)(b0 + row) * TT + t) * OBSD + kt * 32 + kb;
            float4 f0 = *(const float4*)(p);
            float4 f1 = *(const float4*)(p + 4);
            float4 g0 = *(const float4*)(p + 16);
            float4 g1 = *(const float4*)(p + 20);
            v16bf a;
            a[0]  = (__bf16)f0.x; a[1]  = (__bf16)f0.y; a[2]  = (__bf16)f0.z; a[3]  = (__bf16)f0.w;
            a[4]  = (__bf16)f1.x; a[5]  = (__bf16)f1.y; a[6]  = (__bf16)f1.z; a[7]  = (__bf16)f1.w;
            a[8]  = (__bf16)g0.x; a[9]  = (__bf16)g0.y; a[10] = (__bf16)g0.z; a[11] = (__bf16)g0.w;
            a[12] = (__bf16)g1.x; a[13] = (__bf16)g1.y; a[14] = (__bf16)g1.z; a[15] = (__bf16)g1.w;
            if (m == 0) acc0 = wmma_bf16(a, b, acc0);
            else        acc1 = wmma_bf16(a, b, acc1);
          }
        }
        int col = nt * 16 + nlane;
        float bias = (col < NORN) ? in_b[col] : 0.f;
#pragma unroll
        for (int j = 0; j < 8; ++j) {
          sX[(rbase + j) * PORN + col]        = acc0[j] + bias;
          sX[(16 + rbase + j) * PORN + col]   = acc1[j] + bias;
        }
      }
    }
    __syncthreads();

    // ---- 4 inner connectome steps ----
#pragma unroll 1
    for (int it = 0; it < 4; ++it) {
      // -- ORN: ho2 = tanh(hp@Wpto + ho@Woto + hl@Wlto + x + b_orn) --
      v8f nvO[3][2];
#pragma unroll
      for (int ti = 0; ti < 3; ++ti) {
        int nt = wave + ti * 8;
        if (nt < TORN) {
          int col = nt * 16 + nlane;
          v8f a0 = {}, a1 = {};
          gemm_acc2(a0, a1, sPN,  PPN,  KT_PN,  Wpto, TORN, nt, lane);
          gemm_acc2(a0, a1, sORN, PORN, KT_ORN, Woto, TORN, nt, lane);
          gemm_acc2(a0, a1, sLN,  PLN,  KT_LN,  Wlto, TORN, nt, lane);
          float bias = (col < NORN) ? b_orn[col] : 0.f;
#pragma unroll
          for (int j = 0; j < 8; ++j) {
            a0[j] = fast_tanh(a0[j] + sX[(rbase + j) * PORN + col] + bias);
            a1[j] = fast_tanh(a1[j] + sX[(16 + rbase + j) * PORN + col] + bias);
          }
          nvO[ti][0] = a0; nvO[ti][1] = a1;
        }
      }
      __syncthreads();
#pragma unroll
      for (int ti = 0; ti < 3; ++ti) {
        int nt = wave + ti * 8;
        if (nt < TORN) {
          store_tile(sORN, PORN, nt, NORN, nvO[ti][0], lane, 0);
          store_tile(sORN, PORN, nt, NORN, nvO[ti][1], lane, 1);
        }
      }
      __syncthreads();

      // -- PN: hp2 = tanh(ho2@Wotp + hk@Wktp + hl@Wltp + hp@Wptp + b_pn) --
      {
        int nt = wave;  // TPN == 8 tiles, one per wave
        int col = nt * 16 + nlane;
        v8f a0 = {}, a1 = {};
        gemm_acc2(a0, a1, sORN, PORN, KT_ORN, Wotp, TPN, nt, lane);
        gemm_acc2(a0, a1, sKC,  PKC,  KT_KC,  Wktp, TPN, nt, lane);
        gemm_acc2(a0, a1, sLN,  PLN,  KT_LN,  Wltp, TPN, nt, lane);
        gemm_acc2(a0, a1, sPN,  PPN,  KT_PN,  Wptp, TPN, nt, lane);
        float bias = (col < NPN) ? b_pn[col] : 0.f;
#pragma unroll
        for (int j = 0; j < 8; ++j) {
          a0[j] = fast_tanh(a0[j] + bias);
          a1[j] = fast_tanh(a1[j] + bias);
        }
        __syncthreads();
        store_tile(sPN, PPN, nt, NPN, a0, lane, 0);
        store_tile(sPN, PPN, nt, NPN, a1, lane, 1);
        __syncthreads();
      }

      // -- LN: hl2 = tanh(hp2@Wptl + ho2@Wotl + hl@Wltl + b_ln) --
      {
        int nt = wave;  // TLN == 8
        int col = nt * 16 + nlane;
        v8f a0 = {}, a1 = {};
        gemm_acc2(a0, a1, sPN,  PPN,  KT_PN,  Wptl, TLN, nt, lane);
        gemm_acc2(a0, a1, sORN, PORN, KT_ORN, Wotl, TLN, nt, lane);
        gemm_acc2(a0, a1, sLN,  PLN,  KT_LN,  Wltl, TLN, nt, lane);
        float bias = (col < NLN) ? b_ln[col] : 0.f;
#pragma unroll
        for (int j = 0; j < 8; ++j) {
          a0[j] = fast_tanh(a0[j] + bias);
          a1[j] = fast_tanh(a1[j] + bias);
        }
        __syncthreads();
        store_tile(sLN, PLN, nt, NLN, a0, lane, 0);
        store_tile(sLN, PLN, nt, NLN, a1, lane, 1);
        __syncthreads();
      }

      // -- KC: hk2 = tanh(hp2@Wptk + hk@Wktk + b_kc) --
      v8f nvK[4][2];
#pragma unroll
      for (int ti = 0; ti < 4; ++ti) {
        int nt = wave + ti * 8;
        if (nt < TKC) {
          int col = nt * 16 + nlane;
          v8f a0 = {}, a1 = {};
          gemm_acc2(a0, a1, sPN, PPN, KT_PN, Wptk, TKC, nt, lane);
          gemm_acc2(a0, a1, sKC, PKC, KT_KC, Wktk, TKC, nt, lane);
          float bias = (col < NKC) ? b_kc[col] : 0.f;
#pragma unroll
          for (int j = 0; j < 8; ++j) {
            a0[j] = fast_tanh(a0[j] + bias);
            a1[j] = fast_tanh(a1[j] + bias);
          }
          nvK[ti][0] = a0; nvK[ti][1] = a1;
        }
      }
      __syncthreads();
#pragma unroll
      for (int ti = 0; ti < 4; ++ti) {
        int nt = wave + ti * 8;
        if (nt < TKC) {
          store_tile(sKC, PKC, nt, NKC, nvK[ti][0], lane, 0);
          store_tile(sKC, PKC, nt, NKC, nvK[ti][1], lane, 1);
        }
      }
      __syncthreads();
    }

    // ---- heads: mu / clip(log_std) / cast_logit from hk via folded weights ----
    if (tid < MROWS * 5) {
      int row = tid / 5, oc = tid % 5;
      float sum = 0.f;
      for (int k = 0; k < NKC; ++k)
        sum += (float)sKC[row * PKC + k] * headW[k * 5 + oc];
      sum += headB[oc];
      size_t bt = (size_t)(b0 + row) * TT + t;
      if (oc < 2) {
        out[OUT_MU + bt * 2 + oc] = sum;
      } else if (oc < 4) {
        float c = sum < -5.f ? -5.f : (sum > 2.f ? 2.f : sum);
        out[OUT_LS + bt * 2 + (oc - 2)] = c;
      } else {
        out[OUT_C + bt] = sum;
      }
    }
    __syncthreads();
  }

  // ---- h2 = concat(ho, hp, hk, hl) as f32 ----
  for (int idx = tid; idx < 1024; idx += 256) {
    const __bf16* src;
    int stride, c;
    if (idx < NORN)                  { src = sORN; stride = PORN; c = idx; }
    else if (idx < NORN + NPN)       { src = sPN;  stride = PPN;  c = idx - NORN; }
    else if (idx < NORN + NPN + NKC) { src = sKC;  stride = PKC;  c = idx - NORN - NPN; }
    else                             { src = sLN;  stride = PLN;  c = idx - NORN - NPN - NKC; }
    for (int row = 0; row < MROWS; ++row)
      out[OUT_H2 + (size_t)(b0 + row) * 1024 + idx] = (float)src[row * stride + c];
  }
}

// ---------------------------------------------------------------------------
extern "C" void kernel_launch(void* const* d_in, const int* in_sizes, int n_in,
                              void* d_out, int out_size, void* d_ws, size_t ws_size,
                              hipStream_t stream) {
  const float* obs   = (const float*)d_in[0];
  const float* in_W  = (const float*)d_in[1];
  const float* in_b  = (const float*)d_in[2];
  const float* W_pto = (const float*)d_in[3];
  const float* W_oto = (const float*)d_in[4];
  const float* W_lto = (const float*)d_in[5];
  const float* W_otp = (const float*)d_in[6];
  const float* W_ktp = (const float*)d_in[7];
  const float* W_ltp = (const float*)d_in[8];
  const float* W_ptp = (const float*)d_in[9];
  const float* W_ptl = (const float*)d_in[10];
  const float* W_otl = (const float*)d_in[11];
  const float* W_ltl = (const float*)d_in[12];
  const float* W_ptk = (const float*)d_in[13];
  const float* W_ktk = (const float*)d_in[14];
  const float* b_orn = (const float*)d_in[15];
  const float* b_pn  = (const float*)d_in[16];
  const float* b_kc  = (const float*)d_in[17];
  const float* b_ln  = (const float*)d_in[18];
  const float* R_W   = (const float*)d_in[19];
  const float* R_b   = (const float*)d_in[20];
  const float* mu_W  = (const float*)d_in[21];
  const float* mu_b  = (const float*)d_in[22];
  const float* ls_W  = (const float*)d_in[23];
  const float* ls_b  = (const float*)d_in[24];
  const float* c_W   = (const float*)d_in[25];
  const float* c_b   = (const float*)d_in[26];

  __bf16* wp    = (__bf16*)d_ws;
  float*  headW = (float*)((char*)d_ws + O_END * sizeof(__bf16));
  float*  headB = headW + NKC * 5;
  float*  out   = (float*)d_out;

  // pack weights into WMMA-B bf16 fragments (resident in 192MB L2 thereafter)
  pack_w_kernel<<<8 * 22, 256, 0, stream>>>(in_W,  wp + O_INW, OBSD, NORN, TORN);
  pack_w_kernel<<<4 * 22, 256, 0, stream>>>(W_pto, wp + O_PTO, NPN,  NORN, TORN);
  pack_w_kernel<<<11 * 22, 256, 0, stream>>>(W_oto, wp + O_OTO, NORN, NORN, TORN);
  pack_w_kernel<<<4 * 22, 256, 0, stream>>>(W_lto, wp + O_LTO, NLN,  NORN, TORN);
  pack_w_kernel<<<11 * 8, 256, 0, stream>>>(W_otp, wp + O_OTP, NORN, NPN, TPN);
  pack_w_kernel<<<15 * 8, 256, 0, stream>>>(W_ktp, wp + O_KTP, NKC,  NPN, TPN);
  pack_w_kernel<<<4 * 8, 256, 0, stream>>>(W_ltp, wp + O_LTP, NLN,  NPN, TPN);
  pack_w_kernel<<<4 * 8, 256, 0, stream>>>(W_ptp, wp + O_PTP, NPN,  NPN, TPN);
  pack_w_kernel<<<4 * 8, 256, 0, stream>>>(W_ptl, wp + O_PTL, NPN,  NLN, TLN);
  pack_w_kernel<<<11 * 8, 256, 0, stream>>>(W_otl, wp + O_OTL, NORN, NLN, TLN);
  pack_w_kernel<<<4 * 8, 256, 0, stream>>>(W_ltl, wp + O_LTL, NLN,  NLN, TLN);
  pack_w_kernel<<<4 * 30, 256, 0, stream>>>(W_ptk, wp + O_PTK, NPN,  NKC, TKC);
  pack_w_kernel<<<15 * 30, 256, 0, stream>>>(W_ktk, wp + O_KTK, NKC,  NKC, TKC);

  // fold readout (454x1024) through tiny heads into 454x5 — removes largest GEMM
  fold_heads_kernel<<<(NKC * 5 + 5 + 255) / 256, 256, 0, stream>>>(
      R_W, R_b, mu_W, mu_b, ls_W, ls_b, c_W, c_b, headW, headB);

  // persistent recurrence: 32 workgroups x 32 batch rows, state in LDS
  connectome_recurrence<<<BSZ / MROWS, 256, 0, stream>>>(
      wp, headW, headB, obs, in_b, b_orn, b_pn, b_kc, b_ln, out);
}
